// RelativeMultiheadAttention_65996467470625
// MI455X (gfx1250) — matrix operand order repaired and checked by reading, not compiled, not run on hardware
//
#include <hip/hip_runtime.h>
#include <hip/hip_bf16.h>

#define L_DIM 2048
#define S_DIM 2048
#define N_B   4
#define E_DIM 256
#define H_HEADS 8
#define HD    32
#define NH    32   // N_B * H_HEADS

typedef __attribute__((ext_vector_type(16))) __bf16         v16bf;
typedef __attribute__((ext_vector_type(16))) unsigned short v16u;
typedef __attribute__((ext_vector_type(8)))  unsigned short v8u;
typedef __attribute__((ext_vector_type(8)))  float          v8f;

// native f32 -> bf16 conversion (let the backend pick the hardware cvt)
__device__ __forceinline__ unsigned short f2bf(float f) {
    return __builtin_bit_cast(unsigned short, (__bf16)f);
}

__device__ __forceinline__ v16bf castbf(v16u v) { return __builtin_bit_cast(v16bf, v); }

__device__ __forceinline__ v8f vzero8() {
    v8f z;
#pragma unroll
    for (int i = 0; i < 8; ++i) z[i] = 0.0f;
    return z;
}

__device__ __forceinline__ v8f wmma_bf16(v16bf a, v16bf b, v8f c) {
    return __builtin_amdgcn_wmma_f32_16x16x32_bf16(false, a, false, b, (short)0, c, false, false);
}

// ---- fragment loaders -------------------------------------------------------
// A fragment (16 x 32, bf16): lane holds row = lane&15, K = kb..kb+7 and kb+16..kb+23, kb = (lane>>4)*8
__device__ __forceinline__ v16u loadA_u16(const unsigned short* t, int stride) {
    int lane = threadIdx.x & 31;
    int r = lane & 15, kb = (lane >> 4) * 8;
    v8u lo = *(const v8u*)(t + r * stride + kb);
    v8u hi = *(const v8u*)(t + r * stride + kb + 16);
    v16u o;
#pragma unroll
    for (int i = 0; i < 8; ++i) { o[i] = lo[i]; o[i + 8] = hi[i]; }
    return o;
}

// B fragment (32 x 16, bf16) from row-major N x K tile: lane holds col = lane&15,
// K = kb..kb+7 and kb+16..kb+23, kb = (lane>>4)*8
__device__ __forceinline__ v16u loadB_u16(const unsigned short* t, int stride) {
    int lane = threadIdx.x & 31;
    int r = lane & 15, kb = (lane >> 4) * 8;
    v8u lo = *(const v8u*)(t + r * stride + kb);
    v8u hi = *(const v8u*)(t + r * stride + kb + 16);
    v16u o;
#pragma unroll
    for (int i = 0; i < 8; ++i) { o[i] = lo[i]; o[i + 8] = hi[i]; }
    return o;
}

__device__ __forceinline__ v16u loadA_f32(const float* t, int stride) {
    int lane = threadIdx.x & 31;
    int r = lane & 15, kb = (lane >> 4) * 8;
    const float4* p0 = (const float4*)(t + r * stride + kb);
    const float4* p1 = (const float4*)(t + r * stride + kb + 16);
    float4 a0 = p0[0], a1 = p0[1], b0 = p1[0], b1 = p1[1];
    v16u o;
    o[0]  = f2bf(a0.x); o[1]  = f2bf(a0.y); o[2]  = f2bf(a0.z); o[3]  = f2bf(a0.w);
    o[4]  = f2bf(a1.x); o[5]  = f2bf(a1.y); o[6]  = f2bf(a1.z); o[7]  = f2bf(a1.w);
    o[8]  = f2bf(b0.x); o[9]  = f2bf(b0.y); o[10] = f2bf(b0.z); o[11] = f2bf(b0.w);
    o[12] = f2bf(b1.x); o[13] = f2bf(b1.y); o[14] = f2bf(b1.z); o[15] = f2bf(b1.w);
    return o;
}

__device__ __forceinline__ v16u loadB_f32(const float* t, int stride) {
    int lane = threadIdx.x & 31;
    int r = lane & 15, kb = (lane >> 4) * 8;
    const float4* p = (const float4*)(t + r * stride + kb);
    const float4* p2 = (const float4*)(t + r * stride + kb + 16);
    float4 a = p[0], b = p[1], c = p2[0], d = p2[1];
    v16u o;
    o[0]  = f2bf(a.x); o[1]  = f2bf(a.y); o[2]  = f2bf(a.z); o[3]  = f2bf(a.w);
    o[4]  = f2bf(b.x); o[5]  = f2bf(b.y); o[6]  = f2bf(b.z); o[7]  = f2bf(b.w);
    o[8]  = f2bf(c.x); o[9]  = f2bf(c.y); o[10] = f2bf(c.z); o[11] = f2bf(c.w);
    o[12] = f2bf(d.x); o[13] = f2bf(d.y); o[14] = f2bf(d.z); o[15] = f2bf(d.w);
    return o;
}

__device__ __forceinline__ float rmax16(float x) {
#pragma unroll
    for (int m = 1; m < 16; m <<= 1) x = fmaxf(x, __shfl_xor(x, m, 32));
    return x;
}
__device__ __forceinline__ float rsum16(float x) {
#pragma unroll
    for (int m = 1; m < 16; m <<= 1) x += __shfl_xor(x, m, 32);
    return x;
}

// ---- kernel 1: QKV projection (X @ W^T + b), scatter to head-major bf16 ----
// one wave computes a 16M x 64N slab; waves = 3 proj * 512 Mtiles * 4 Ngroups = 6144
__global__ __launch_bounds__(128) void qkv_kernel(
    const float* __restrict__ q, const float* __restrict__ k, const float* __restrict__ v,
    const float* __restrict__ W, const float* __restrict__ bias,
    unsigned short* __restrict__ Qb, unsigned short* __restrict__ Kb, unsigned short* __restrict__ Vt) {
    int lane = threadIdx.x & 31;
    int wid  = blockIdx.x * 4 + (threadIdx.x >> 5);
    int proj = wid >> 11;
    int rem  = wid & 2047;
    int mtile = rem >> 2, ngrp = rem & 3;

    const float* X  = (proj == 0) ? q : ((proj == 1) ? k : v);
    const float* Wp = W + proj * E_DIM * E_DIM;

    v8f acc[4];
#pragma unroll
    for (int g = 0; g < 4; ++g) acc[g] = vzero8();

    for (int kc = 0; kc < E_DIM; kc += 32) {
        v16bf a = castbf(loadA_f32(X + (size_t)(mtile * 16) * E_DIM + kc, E_DIM));
#pragma unroll
        for (int g = 0; g < 4; ++g) {
            v16bf b = castbf(loadB_f32(Wp + (size_t)(ngrp * 64 + g * 16) * E_DIM + kc, E_DIM));
            acc[g] = wmma_bf16(a, b, acc[g]);
        }
    }

    const float scaling = 0.17677669529663687f;  // HD^-0.5
    int half8 = (lane >> 4) * 8;
#pragma unroll
    for (int g = 0; g < 4; ++g) {
        int col = ngrp * 64 + g * 16 + (lane & 15);
        float bv = bias[proj * E_DIM + col];
        int h = col >> 5, d = col & 31;
#pragma unroll
        for (int j = 0; j < 8; ++j) {
            int m  = mtile * 16 + half8 + j;   // row of flattened (seq*N, E)
            int li = m >> 2, ni = m & 3;
            int bh = ni * H_HEADS + h;         // head-batch index
            float val = acc[g][j] + bv;
            if (proj == 0)      Qb[((size_t)bh * L_DIM + li) * HD + d] = f2bf(val * scaling);
            else if (proj == 1) Kb[((size_t)bh * S_DIM + li) * HD + d] = f2bf(val);
            else                Vt[((size_t)bh * HD + d) * S_DIM + li] = f2bf(val);
        }
    }
}

// online-softmax update for one 16-row group over 32 new columns
__device__ __forceinline__ void online_update(
    v8f& c0, v8f& c1, float* mrun, float* lrun, v8f& o0, v8f& o1,
    unsigned short* pbuf, int half8, int c16) {
    const float LOG2E = 1.44269504088896f;
#pragma unroll
    for (int j = 0; j < 8; ++j) {
        float tm = rmax16(fmaxf(c0[j], c1[j]));
        float mn = fmaxf(mrun[j], tm);
        float al = exp2f((mrun[j] - mn) * LOG2E);
        float p0 = exp2f((c0[j] - mn) * LOG2E);
        float p1 = exp2f((c1[j] - mn) * LOG2E);
        lrun[j] = lrun[j] * al + rsum16(p0 + p1);
        mrun[j] = mn;
        o0[j] *= al;
        o1[j] *= al;
        pbuf[(half8 + j) * 32 + c16]      = f2bf(p0);
        pbuf[(half8 + j) * 32 + 16 + c16] = f2bf(p1);
    }
}

// ---- kernel 2: flash attention with additive bias ---------------------------
// one wave per (head, 32-row L tile); 32*64 = 2048 waves
__global__ __launch_bounds__(128) void attn_kernel(
    const unsigned short* __restrict__ Qb, const unsigned short* __restrict__ Kb,
    const unsigned short* __restrict__ Vt, const float* __restrict__ rel,
    unsigned short* __restrict__ Obuf) {
    __shared__ __align__(16) unsigned short plds[4][1024];   // per-wave 32x32 bf16 staging
    int lane = threadIdx.x & 31;
    int warp = threadIdx.x >> 5;
    int wid  = blockIdx.x * 4 + warp;
    int b  = wid >> 6;
    int lt = wid & 63;

    const unsigned short* Qbase = Qb + ((size_t)b * L_DIM + lt * 32) * HD;
    v16bf qf0 = castbf(loadA_u16(Qbase, HD));
    v16bf qf1 = castbf(loadA_u16(Qbase + 16 * HD, HD));

    float mrun[16], lrun[16];
#pragma unroll
    for (int j = 0; j < 16; ++j) { mrun[j] = -3.0e38f; lrun[j] = 0.0f; }
    v8f o00 = vzero8(), o01 = vzero8(), o10 = vzero8(), o11 = vzero8();

    const float* relb = rel + (size_t)b * L_DIM * S_DIM + (size_t)(lt * 32) * S_DIM;
    int half8 = (lane >> 4) * 8, c16 = lane & 15;

    for (int s0 = 0; s0 < S_DIM; s0 += 32) {
        // prefetch the bias stream 4 chunks (128B/row) ahead: lane i -> row i of this slab
        if (s0 + 128 < S_DIM)
            __builtin_prefetch(relb + (size_t)lane * S_DIM + s0 + 128, 0, 1);

        // bias tiles as WMMA C accumulators; single-use 512MB stream -> non-temporal
        v8f c00, c01, c10, c11;
#pragma unroll
        for (int j = 0; j < 8; ++j) {
            const float* rp = relb + (size_t)(half8 + j) * S_DIM + s0 + c16;
            const float* rq = rp + (size_t)16 * S_DIM;
            c00[j] = __builtin_nontemporal_load(rp);
            c01[j] = __builtin_nontemporal_load(rp + 16);
            c10[j] = __builtin_nontemporal_load(rq);
            c11[j] = __builtin_nontemporal_load(rq + 16);
        }
        v16bf k0 = castbf(loadB_u16(Kb + ((size_t)b * S_DIM + s0) * HD, HD));
        v16bf k1 = castbf(loadB_u16(Kb + ((size_t)b * S_DIM + s0 + 16) * HD, HD));
        c00 = wmma_bf16(qf0, k0, c00);   // scores = Q K^T + bias (Q pre-scaled)
        c01 = wmma_bf16(qf0, k1, c01);
        c10 = wmma_bf16(qf1, k0, c10);
        c11 = wmma_bf16(qf1, k1, c11);

        online_update(c00, c01, mrun,     lrun,     o00, o01, &plds[warp][0],   half8, c16);
        online_update(c10, c11, mrun + 8, lrun + 8, o10, o11, &plds[warp][512], half8, c16);

        v16bf pf0 = castbf(loadA_u16(&plds[warp][0],   32));
        v16bf pf1 = castbf(loadA_u16(&plds[warp][512], 32));
        v16bf v0  = castbf(loadB_u16(Vt + (size_t)(b * HD) * S_DIM + s0,      S_DIM));
        v16bf v1  = castbf(loadB_u16(Vt + (size_t)(b * HD + 16) * S_DIM + s0, S_DIM));
        o00 = wmma_bf16(pf0, v0, o00);
        o01 = wmma_bf16(pf0, v1, o01);
        o10 = wmma_bf16(pf1, v0, o10);
        o11 = wmma_bf16(pf1, v1, o11);
    }

    int nb = b >> 3, h = b & 7;
#pragma unroll
    for (int j = 0; j < 8; ++j) {
        float inv0 = 1.0f / lrun[j];
        float inv1 = 1.0f / lrun[8 + j];
        int row0 = lt * 32 + half8 + j;
        int row1 = row0 + 16;
        unsigned short* op0 = Obuf + ((size_t)row0 * N_B + nb) * E_DIM + h * HD;
        unsigned short* op1 = Obuf + ((size_t)row1 * N_B + nb) * E_DIM + h * HD;
        op0[c16]      = f2bf(o00[j] * inv0);
        op0[16 + c16] = f2bf(o01[j] * inv0);
        op1[c16]      = f2bf(o10[j] * inv1);
        op1[16 + c16] = f2bf(o11[j] * inv1);
    }
}

// ---- kernel 3: output projection (O @ Wo^T + bo) -> fp32 --------------------
// 512 Mtiles * 4 Ngroups = 2048 waves
__global__ __launch_bounds__(128) void oproj_kernel(
    const unsigned short* __restrict__ Obuf, const float* __restrict__ Wo,
    const float* __restrict__ bo, float* __restrict__ out) {
    int lane = threadIdx.x & 31;
    int wid  = blockIdx.x * 4 + (threadIdx.x >> 5);
    int mtile = wid >> 2, ngrp = wid & 3;

    v8f acc[4];
#pragma unroll
    for (int g = 0; g < 4; ++g) acc[g] = vzero8();

    for (int kc = 0; kc < E_DIM; kc += 32) {
        v16bf a = castbf(loadA_u16(Obuf + (size_t)(mtile * 16) * E_DIM + kc, E_DIM));
#pragma unroll
        for (int g = 0; g < 4; ++g) {
            v16bf b = castbf(loadB_f32(Wo + (size_t)(ngrp * 64 + g * 16) * E_DIM + kc, E_DIM));
            acc[g] = wmma_bf16(a, b, acc[g]);
        }
    }

    int half8 = (lane >> 4) * 8;
#pragma unroll
    for (int g = 0; g < 4; ++g) {
        int col = ngrp * 64 + g * 16 + (lane & 15);
        float bv = bo[col];
#pragma unroll
        for (int j = 0; j < 8; ++j) {
            out[(size_t)(mtile * 16 + half8 + j) * E_DIM + col] = acc[g][j] + bv;
        }
    }
}

extern "C" void kernel_launch(void* const* d_in, const int* in_sizes, int n_in,
                              void* d_out, int out_size, void* d_ws, size_t ws_size,
                              hipStream_t stream) {
    const float* query = (const float*)d_in[0];
    const float* key   = (const float*)d_in[1];
    const float* value = (const float*)d_in[2];
    const float* rel   = (const float*)d_in[3];
    const float* Wqkv  = (const float*)d_in[4];
    const float* bqkv  = (const float*)d_in[5];
    const float* Wo    = (const float*)d_in[6];
    const float* bo    = (const float*)d_in[7];
    float* out = (float*)d_out;

    const size_t HBUF = (size_t)NH * S_DIM * HD;   // 2,097,152 bf16 elems = 4MB
    unsigned short* Qb   = (unsigned short*)d_ws;
    unsigned short* Kb   = Qb + HBUF;
    unsigned short* Vt   = Kb + HBUF;
    unsigned short* Obuf = Vt + HBUF;              // (L*N, E) bf16, 4MB

    qkv_kernel<<<1536, 128, 0, stream>>>(query, key, value, Wqkv, bqkv, Qb, Kb, Vt);
    attn_kernel<<<512, 128, 0, stream>>>(Qb, Kb, Vt, rel, Obuf);
    oproj_kernel<<<512, 128, 0, stream>>>(Obuf, Wo, bo, out);
}
